// RNN_Separate_39822936769234
// MI455X (gfx1250) — matrix operand order repaired and checked by reading, not compiled, not run on hardware
//
#include <hip/hip_runtime.h>
#include <cstdint>
#include <cstddef>

// Problem sizes (fixed by the reference).
#define B_SZ   64
#define T_SZ   1024
#define NI_SZ  128
#define NH_SZ  512

// Phase-2 persistent config.
#define P2_NBLK   32                       // 32 blocks x 16 h-columns = NH
#define P2_PITCH  516                      // dwords per logical 512-dword LDS row (TDM pad: +2dw per 256dw)
#define P2_SMEM   ((64 * P2_PITCH + 16 * P2_PITCH) * 4)   // state (64 rows) + Wh slice (16 rows) = 165,120 B

typedef __attribute__((ext_vector_type(16))) __bf16 v16bf;
typedef __attribute__((ext_vector_type(4)))  __bf16 v4bf;
typedef __attribute__((ext_vector_type(8)))  float  v8f;
typedef __attribute__((ext_vector_type(4)))  float  v4f;
typedef __attribute__((ext_vector_type(2)))  float  v2f;
typedef __attribute__((ext_vector_type(4)))  unsigned u32x4;
typedef __attribute__((ext_vector_type(8)))  int     i32x8;
typedef __attribute__((ext_vector_type(4)))  int     i32x4;

#if defined(__HIP_DEVICE_COMPILE__)
#  if __has_builtin(__builtin_amdgcn_tensor_load_to_lds)
#    define HAVE_TDM 1
#  else
#    define HAVE_TDM 0
#  endif
#  if __has_builtin(__builtin_amdgcn_tanhf)
#    define HAVE_TANH 1
#  else
#    define HAVE_TANH 0
#  endif
#else
#  define HAVE_TDM 0
#  define HAVE_TANH 0
#endif

// ---------------- helpers ----------------

// Convert 4 f32 -> 4 bf16 (two packed dwords). Let ISel pick native packed cvt.
__device__ __forceinline__ uint2 cvt4_bf16(float4 f) {
  const v4f v = {f.x, f.y, f.z, f.w};
  const v4bf b = __builtin_convertvector(v, v4bf);
  return __builtin_bit_cast(uint2, b);
}

// Native V_TANH_F32 (gfx1250 transcendental) when available.
__device__ __forceinline__ float fast_tanh(float x) {
#if HAVE_TANH
  return __builtin_amdgcn_tanhf(x);
#else
  return tanhf(x);
#endif
}

// logical k (0..511) -> padded LDS dword index (TDM inserts 2 dwords after every 256)
__device__ __forceinline__ int chunked(int k) { return ((k >> 8) * 258) + (k & 255); }

struct BfPack8   { unsigned u[8]; };   // 32B -> v16bf
struct BfPack2x4 { uint4 a, b; };      // 32B -> v16bf

// TDM: DMA a (tile_rows x 512) f32 tile, row stride 512, into LDS with a
// 516-dword padded row pitch (D# pad_enable, interval=256dw, amount=2dw).
__device__ __forceinline__ void tdm_load_tile_f32(float* lds, const float* gsrc,
                                                  int tile_rows, int tensor_rows) {
#if HAVE_TDM
  const unsigned lds_off = (unsigned)(uintptr_t)lds;               // low 32 bits of flat addr = LDS offset
  const unsigned long long ga = (unsigned long long)(uintptr_t)gsrc;
  const u32x4 g0 = { 1u,                                           // count=1, gather off
                     lds_off,                                      // lds_addr
                     (unsigned)(ga & 0xFFFFFFFFu),                 // global_addr[31:0]
                     (unsigned)((ga >> 32) & 0x1FFFFFFu) | (2u << 30) }; // addr[56:32] | type=2
  const i32x8 g1 = {
      (int)((2u << 16) | (1u << 20) | (7u << 22) | (1u << 25)),    // data_size=4B, pad_en, 256dw interval, 2dw pad
      (int)(512u << 16),                                           // tensor_dim0 = 512
      (int)(((unsigned)tensor_rows & 0xFFFFu) << 16),              // tensor_dim1
      (int)(512u << 16),                                           // tile_dim0 = 512
      (int)((unsigned)tile_rows & 0xFFFFu),                        // tile_dim1 (tile_dim2 = 0)
      512,                                                         // tensor_dim0_stride
      0, 0 };
  const i32x4 z4 = { 0, 0, 0, 0 };
#if __clang_major__ >= 23
  const i32x8 z8 = { 0, 0, 0, 0, 0, 0, 0, 0 };
  __builtin_amdgcn_tensor_load_to_lds(g0, g1, z4, z4, z8, 0);
#else
  __builtin_amdgcn_tensor_load_to_lds(g0, g1, z4, z4, 0);
#endif
  __builtin_amdgcn_s_wait_tensorcnt(0);
#else
  // Fallback: calling wave loads cooperatively.
  const int lane = (int)(threadIdx.x & 31);
  for (int r = 0; r < tile_rows; ++r)
    for (int c = lane; c < 512; c += 32)
      lds[r * P2_PITCH + chunked(c)] = gsrc[(size_t)r * 512 + c];
  (void)tensor_rows;
#endif
}

// Cumulative grid barrier across P2_NBLK persistent blocks.
__device__ __forceinline__ void grid_barrier(unsigned* ctr, unsigned target) {
  __threadfence();                 // release: every thread's stores to device scope
  __syncthreads();
  if (threadIdx.x == 0) {
    atomicAdd(ctr, 1u);
    while (__hip_atomic_load(ctr, __ATOMIC_RELAXED, __HIP_MEMORY_SCOPE_AGENT) < target)
      __builtin_amdgcn_s_sleep(2);
    __threadfence();               // acquire: invalidate stale near caches
  }
  __syncthreads();
}

// ---------------- phase 0: zero the barrier counter (graph-replay safe) ----------------

__global__ void rnn455_init(unsigned* ctr) {
  if (threadIdx.x == 0)
    __hip_atomic_store(ctr, 0u, __ATOMIC_RELAXED, __HIP_MEMORY_SCOPE_AGENT);
}

// ---------------- phase 1: u = tanh(x @ Wi^T + bi) -> written into d_out[b][t][h] ----------------
// grid (1024, 4), 256 threads (8 waves). Block: 64 rows x 128 cols; wave w owns 16 cols, 4 row-tiles.
// bf16 WMMA 16x16x32, K = NI = 128 (4 WMMA per tile).

__global__ __launch_bounds__(256)
void rnn455_phase1(const float* __restrict__ x, const float* __restrict__ Wi,
                   const float* __restrict__ bi, float* __restrict__ out) {
  __shared__ unsigned char As[64 * 272];   // 64 rows x 128 bf16, 272B pitch (16B pad)
  const int tid = (int)threadIdx.x;
  const int r0  = (int)blockIdx.x * 64;

  // Stage A tile (x rows) into LDS as bf16. 2048 float4 total, 8 per thread.
  for (int j = tid; j < 64 * 32; j += 256) {
    const int row = j >> 5;
    const int c4  = (j & 31) << 2;                       // float index within row
    const float4 f = *(const float4*)(x + (size_t)(r0 + row) * NI_SZ + c4);
    *(uint2*)(As + row * 272 + c4 * 2) = cvt4_bf16(f);
  }
  __syncthreads();

  const int wave = tid >> 5, lane = tid & 31;
  const int n  = lane & 15;        // A-row / B-col within tile
  const int hf = lane >> 4;        // lane half (K split)
  const int hcol = (int)blockIdx.y * 128 + wave * 16 + n;

  // Preload B (Wi columns) straight from global (L2-resident), converting to bf16.
  // B layout: lane half hf covers K = kk*32 + 16*hf .. +15 (16 contiguous elements).
  v16bf breg[4];
#pragma unroll
  for (int kk = 0; kk < 4; ++kk) {
    const float* wp = Wi + (size_t)hcol * NI_SZ + kk * 32 + hf * 16;
    BfPack8 u;
#pragma unroll
    for (int q = 0; q < 4; ++q) {
      const float4 f = *(const float4*)(wp + q * 4);
      const uint2 c = cvt4_bf16(f);
      u.u[q * 2]     = c.x;
      u.u[q * 2 + 1] = c.y;
    }
    breg[kk] = __builtin_bit_cast(v16bf, u);
  }
  const float biv = bi[hcol];

#pragma unroll
  for (int mt = 0; mt < 4; ++mt) {
    v8f acc = {0.f, 0.f, 0.f, 0.f, 0.f, 0.f, 0.f, 0.f};
#pragma unroll
    for (int kk = 0; kk < 4; ++kk) {
      // A layout: lane row = n; K = kk*32 + 16*hf..+15 contiguous (two 16B LDS reads).
      const unsigned char* base = As + (mt * 16 + n) * 272 + (kk * 32 + hf * 16) * 2;
      BfPack2x4 pr;
      pr.a = *(const uint4*)(base);
      pr.b = *(const uint4*)(base + 16);
      const v16bf a = __builtin_bit_cast(v16bf, pr);
      acc = __builtin_amdgcn_wmma_f32_16x16x32_bf16(false, a, false, breg[kk],
                                                    (short)0, acc, false, false);
    }
    // C layout: VGPR i -> row (i + 8*hf), col n.
#pragma unroll
    for (int i = 0; i < 8; ++i) {
      const int row = r0 + mt * 16 + i + hf * 8;
      out[(size_t)row * NH_SZ + hcol] = fast_tanh(acc[i] + biv);
    }
  }
}

// ---------------- phase 2: serial recurrence, 32 persistent blocks x 128 threads ----------------
// state_t = u_t + tanh(state_{t-1} @ Wh^T + bh), written in-place over u in d_out.
// f32 WMMA 16x16x4, K = NH = 512, 4 interleaved accumulators to break the D->C chain.

__global__ __launch_bounds__(128)
void rnn455_phase2(const float* __restrict__ Wh, const float* __restrict__ bh,
                   float* __restrict__ out, float* __restrict__ sbuf,
                   unsigned* __restrict__ ctr) {
  extern __shared__ float smem[];
  float* Sf  = smem;                    // 64 x 516 dwords: prev state, fp32
  float* Whs = smem + 64 * P2_PITCH;    // 16 x 516 dwords: Wh slice, fp32

  const int tid  = (int)threadIdx.x;
  const int wave = tid >> 5, lane = tid & 31;
  const int n  = lane & 15;             // A-row / B-col within tile
  const int hf = lane >> 4;             // lane half (K split)
  const int h0 = (int)blockIdx.x * 16;
  const int hcol = h0 + n;
  const int m0 = wave * 16;             // this wave's 16 batch rows

  // One-time: TDM the Wh slice (rows h0..h0+15) into LDS.
  if (wave == 0)
    tdm_load_tile_f32(Whs, Wh + (size_t)h0 * NH_SZ, 16, NH_SZ);
  const float bhv = bh[hcol];
  __syncthreads();

  for (int t = 0; t < T_SZ; ++t) {
    // Stage previous state into LDS.
    if (t == 0) {
      for (int j = tid; j < 64 * P2_PITCH; j += 128) Sf[j] = 0.0f;   // init state = 0
    } else {
      grid_barrier(ctr, (unsigned)(P2_NBLK * t));                    // step t-1 fully published
      if (wave == 0)
        tdm_load_tile_f32(Sf, sbuf + (size_t)((t - 1) & 1) * (64 * NH_SZ), 64, 64);
    }
    __syncthreads();

    const v8f z = {0.f, 0.f, 0.f, 0.f, 0.f, 0.f, 0.f, 0.f};
    v8f acc[4] = {z, z, z, z};
#pragma unroll 4
    for (int kb = 0; kb < NH_SZ; kb += 16) {
#pragma unroll
      for (int j = 0; j < 4; ++j) {
        const int k = kb + 4 * j + 2 * hf;            // lane covers K = k, k+1
        const v2f av = *(const v2f*)(Sf  + (m0 + n) * P2_PITCH + chunked(k));
        const v2f bv = *(const v2f*)(Whs + n        * P2_PITCH + chunked(k));
        acc[j] = __builtin_amdgcn_wmma_f32_16x16x4_f32(false, av, false, bv,
                                                       (short)0, acc[j], false, false);
      }
    }
    const v8f accv = (acc[0] + acc[1]) + (acc[2] + acc[3]);

    // Epilogue: read u from d_out, overwrite with new state, publish to ping-pong buffer.
#pragma unroll
    for (int i = 0; i < 8; ++i) {
      const int bg = m0 + i + hf * 8;                 // C layout row
      const size_t oidx = (size_t)bg * (T_SZ * NH_SZ) + (size_t)t * NH_SZ + hcol;
      const float v = out[oidx] + fast_tanh(accv[i] + bhv);
      out[oidx] = v;
      sbuf[(size_t)(t & 1) * (64 * NH_SZ) + (size_t)bg * NH_SZ + hcol] = v;
      if (t == T_SZ - 1)
        out[(size_t)B_SZ * T_SZ * NH_SZ + (size_t)bg * NH_SZ + hcol] = v;  // `last` output
    }
    __syncthreads();
  }
}

// ---------------- launcher ----------------

extern "C" void kernel_launch(void* const* d_in, const int* in_sizes, int n_in,
                              void* d_out, int out_size, void* d_ws, size_t ws_size,
                              hipStream_t stream) {
  (void)in_sizes; (void)n_in; (void)out_size; (void)ws_size;
  const float* x  = (const float*)d_in[0];
  const float* Wi = (const float*)d_in[1];
  const float* bi = (const float*)d_in[2];
  const float* Wh = (const float*)d_in[3];
  const float* bh = (const float*)d_in[4];
  float*    out  = (float*)d_out;
  unsigned* ctr  = (unsigned*)d_ws;                          // barrier counter
  float*    sbuf = (float*)((char*)d_ws + 256);              // 2 x [64 x 512] f32 state ping-pong

  hipFuncSetAttribute((const void*)rnn455_phase2,
                      hipFuncAttributeMaxDynamicSharedMemorySize, P2_SMEM);

  rnn455_init<<<dim3(1), dim3(32), 0, stream>>>(ctr);
  rnn455_phase1<<<dim3((B_SZ * T_SZ) / 64, NH_SZ / 128), dim3(256), 0, stream>>>(x, Wi, bi, out);
  rnn455_phase2<<<dim3(P2_NBLK), dim3(128), P2_SMEM, stream>>>(Wh, bh, out, sbuf, ctr);
}
